// QuantumAttention_65481071395348
// MI455X (gfx1250) — matrix operand order repaired and checked by reading, not compiled, not run on hardware
//
#include <hip/hip_runtime.h>

typedef _Float16 v16h __attribute__((ext_vector_type(16)));
typedef _Float16 v8h  __attribute__((ext_vector_type(8)));
typedef float    v8f  __attribute__((ext_vector_type(8)));
typedef float    v4f  __attribute__((ext_vector_type(4)));

#define K_TOTAL 512
#define N_TOTAL 512
#define BM      64          // M-rows per workgroup
#define BK      32          // K per step (matches wmma_f32_16x16x32_f16)
#define KSTEPS  (K_TOTAL / BK)
#define LDS_STRIDE 40       // f16 per A-tile row (80 B, padded, 16B-aligned)

// ---------------------------------------------------------------------------
// Prep: convert W (f32 [N,K] row-major) to f16, same layout. Wh[n*K + k].
// ---------------------------------------------------------------------------
__global__ void w_to_half_kernel(const float* __restrict__ W,
                                 _Float16* __restrict__ Wh, int n) {
  int i = blockIdx.x * blockDim.x + threadIdx.x;
  if (i < n) Wh[i] = (_Float16)W[i];
}

// ---------------------------------------------------------------------------
// Main: out[m,n] = sum_k cos(x[m,k]+phi[k&7]) * W[n,k] + b[n]
// 256 threads = 8 waves. Wave w computes rows [m0, m0+64) x cols [64w, 64w+64).
// ---------------------------------------------------------------------------
__global__ __launch_bounds__(256)
void qattn_gemm_kernel(const float* __restrict__ x,
                       const float* __restrict__ phi,
                       const _Float16* __restrict__ Wh,
                       const float* __restrict__ bias,
                       float* __restrict__ out) {
  __shared__ _Float16 Atile[BM * LDS_STRIDE];

  const int tid      = threadIdx.x;
  const int lane     = tid & 31;
  const int wave     = tid >> 5;       // 0..7
  const int laneMod  = lane & 15;      // column / row within 16
  const int laneHalf = lane >> 4;      // 0 or 1
  const int m0       = blockIdx.x * BM;
  const int n0       = wave * 64;

  // phi: 8 trainable angles, index = k % 8
  float p[8];
#pragma unroll
  for (int j = 0; j < 8; ++j) p[j] = phi[j];

  // A-staging map: thread t loads 8 consecutive K of one row.
  const int arow = tid >> 2;           // 0..63
  const int acol = (tid & 3) * 8;      // 0,8,16,24  (phi idx == j since acol%8==0)
  const float* xp = x + (size_t)(m0 + arow) * K_TOTAL + acol;

  v8f acc[4][4];
#pragma unroll
  for (int i = 0; i < 4; ++i)
#pragma unroll
    for (int j = 0; j < 4; ++j)
#pragma unroll
      for (int r = 0; r < 8; ++r) acc[i][j][r] = 0.0f;

  // prefetch first x chunk into registers
  v4f xa = *(const v4f*)(xp);
  v4f xb = *(const v4f*)(xp + 4);

  for (int ks = 0; ks < KSTEPS; ++ks) {
    __syncthreads();                   // previous-step LDS reads complete
    // cos + f16 pack -> LDS
    v8h ah;
#pragma unroll
    for (int j = 0; j < 4; ++j) ah[j]     = (_Float16)__cosf(xa[j] + p[j]);
#pragma unroll
    for (int j = 0; j < 4; ++j) ah[4 + j] = (_Float16)__cosf(xb[j] + p[4 + j]);
    *(v8h*)(&Atile[arow * LDS_STRIDE + acol]) = ah;
    __syncthreads();                   // A-tile visible to all waves

    // prefetch next x chunk (overlaps with WMMA phase)
    if (ks + 1 < KSTEPS) {
      xa = *(const v4f*)(xp + (ks + 1) * BK);
      xb = *(const v4f*)(xp + (ks + 1) * BK + 4);
    }

    // B fragments straight from L2 (f16 W, row-major [n][k]):
    // lane laneMod holds column n0+16j+laneMod, K = ks*32 + laneHalf*8 + {0-7, 16-23}
    v16h bfrag[4];
#pragma unroll
    for (int j = 0; j < 4; ++j) {
      const _Float16* bp = Wh + (size_t)(n0 + 16 * j + laneMod) * K_TOTAL
                              + ks * BK + laneHalf * 8;
      bfrag[j] = __builtin_shufflevector(*(const v8h*)bp, *(const v8h*)(bp + 16),
                                         0, 1, 2, 3, 4, 5, 6, 7, 8, 9, 10, 11, 12, 13, 14, 15);
      if (ks + 1 < KSTEPS) __builtin_prefetch(bp + BK, 0, 1);
    }

    // A fragments from LDS:
    // lanes 0-15:  K chunks {0-7, 16-23}; lanes 16-31: {8-15, 24-31}
    v16h afrag[4];
#pragma unroll
    for (int i = 0; i < 4; ++i) {
      const _Float16* ap = &Atile[(16 * i + laneMod) * LDS_STRIDE + laneHalf * 8];
      v8h lo = *(const v8h*)(ap);
      v8h hi = *(const v8h*)(ap + 16);
      afrag[i] = __builtin_shufflevector(lo, hi, 0, 1, 2, 3, 4, 5, 6, 7,
                                                 8, 9, 10, 11, 12, 13, 14, 15);
    }

    // 16 WMMAs: 64x64 patch advanced by K=32
#pragma unroll
    for (int i = 0; i < 4; ++i)
#pragma unroll
      for (int j = 0; j < 4; ++j)
        acc[i][j] = __builtin_amdgcn_wmma_f32_16x16x32_f16(
            false, afrag[i], false, bfrag[j], (short)0, acc[i][j], false, false);
  }

  // Epilogue: bias add + store (VGPR r -> M = 16i + r + 8*laneHalf, N = 16j+laneMod)
#pragma unroll
  for (int j = 0; j < 4; ++j) {
    const int ncol = n0 + 16 * j + laneMod;
    const float bb = bias[ncol];
#pragma unroll
    for (int i = 0; i < 4; ++i) {
      float* op = out + (size_t)(m0 + 16 * i + 8 * laneHalf) * N_TOTAL + ncol;
#pragma unroll
      for (int r = 0; r < 8; ++r)
        op[(size_t)r * N_TOTAL] = acc[i][j][r] + bb;
    }
  }
}

// ---------------------------------------------------------------------------
extern "C" void kernel_launch(void* const* d_in, const int* in_sizes, int n_in,
                              void* d_out, int out_size, void* d_ws, size_t ws_size,
                              hipStream_t stream) {
  const float* x    = (const float*)d_in[0];  // [8,8192,512] f32
  const float* phi  = (const float*)d_in[1];  // [8] f32
  const float* W    = (const float*)d_in[2];  // [512,512] f32
  const float* bias = (const float*)d_in[3];  // [512] f32
  float* out        = (float*)d_out;          // [8,8192,512] f32
  _Float16* Wh      = (_Float16*)d_ws;        // 512 KB f16 copy of W

  const int wn = N_TOTAL * K_TOTAL;
  w_to_half_kernel<<<(wn + 255) / 256, 256, 0, stream>>>(W, Wh, wn);

  const int M = 8 * 8192;                     // 65536 rows
  qattn_gemm_kernel<<<M / BM, 256, 0, stream>>>(x, phi, Wh, bias, out);
}
